// DTransformerLayer_7017976562042
// MI455X (gfx1250) — matrix operand order repaired and checked by reading, not compiled, not run on hardware
//
#include <hip/hip_runtime.h>
#include <hip/hip_bf16.h>
#include <math.h>

// B=4, S=1024, D=1024, H=16, DK=64
constexpr int Bb = 4, S = 1024, D = 1024, H = 16, DK = 64;
constexpr int M = Bb * S; // 4096 rows for all GEMMs

typedef _Float16 h8  __attribute__((ext_vector_type(8)));
typedef _Float16 h16 __attribute__((ext_vector_type(16)));
typedef float    f8  __attribute__((ext_vector_type(8)));

union H16 { h16 v; h8 h[2]; };

__device__ __forceinline__ f8 wmma16(const h16& a, const h16& b, const f8& c) {
  return __builtin_amdgcn_wmma_f32_16x16x32_f16(false, a, false, b, (short)0, c,
                                                false, false);
}

enum { MODE_HEAD = 0, MODE_VT = 1, MODE_Y = 2 };

// ---------------------------------------------------------------------------
// Tiled WMMA GEMM: C[4096,1024] = A[4096,1024] * W[1024,1024] + bias (+resid)
// Block: 256 thr (8 waves), tile 128x128, K-step 32. Wave grid 4x2 -> each
// wave computes a 32x64 region = 2x4 subtiles of 16x16 via v_wmma f16.
// A_F16 path stages the A tile with global_load_async_to_lds_b128 (ASYNCcnt).
// ---------------------------------------------------------------------------
template <int MODE, bool A_F16>
__global__ __launch_bounds__(256) void gemm_wmma(const void* __restrict__ Aptr,
                                                 const float* __restrict__ W,
                                                 const float* __restrict__ bias,
                                                 const float* __restrict__ resid,
                                                 void* __restrict__ outp,
                                                 float scale) {
  __shared__ _Float16 As[128][40];   // [m][k], pad to 40 (80B rows, 16B-aligned)
  __shared__ _Float16 Bs[128][40];   // [n][k] (transposed W tile)

  const int tid  = threadIdx.x;
  const int lane = tid & 31;
  const int wid  = tid >> 5;
  const int g    = lane >> 4;   // lane half
  const int ln   = lane & 15;
  const int bn   = blockIdx.x;  // 8 tiles over N
  const int bm   = blockIdx.y;  // 32 tiles over M

  const int mBase = (wid >> 1) * 32;  // 0..96
  const int nBase = (wid & 1) * 64;   // 0,64

  f8 acc[2][4];
#pragma unroll
  for (int mi = 0; mi < 2; mi++)
#pragma unroll
    for (int ni = 0; ni < 4; ni++)
#pragma unroll
      for (int r = 0; r < 8; r++) acc[mi][ni][r] = 0.0f;

  const int arow  = tid >> 1;          // 0..127
  const int ahalf = (tid & 1) * 16;    // k offset 0/16
  const int bcol  = tid & 127;         // n within tile
  const int bkg   = (tid >> 7) * 16;   // k half 0/16

  for (int k0 = 0; k0 < D; k0 += 32) {
    // ---- stage A tile ----
    {
      const long r = (long)bm * 128 + arow;
      if (A_F16) {
        // pure f16 copy: async DMA straight into LDS, no VGPR round-trip.
        const _Float16* src = ((const _Float16*)Aptr) + r * D + k0 + ahalf;
        unsigned l0 = (unsigned)(uintptr_t)&As[arow][ahalf];  // LDS byte offset
        asm volatile("global_load_async_to_lds_b128 %0, %1, off"
                     :: "v"(l0), "v"(src) : "memory");
        asm volatile("global_load_async_to_lds_b128 %0, %1, off offset:16"
                     :: "v"(l0), "v"(src) : "memory");
      } else {
        const float* src = ((const float*)Aptr) + r * D + k0 + ahalf;
        h8 h0, h1;
#pragma unroll
        for (int i = 0; i < 8; i++) {
          h0[i] = (_Float16)src[i];
          h1[i] = (_Float16)src[i + 8];
        }
        *(h8*)&As[arow][ahalf]     = h0;
        *(h8*)&As[arow][ahalf + 8] = h1;
      }
    }
    // ---- stage W tile transposed: Bs[n][k]; coalesced 4B loads per k-row ----
    {
#pragma unroll
      for (int kk = 0; kk < 16; kk++) {
        float v = W[(long)(k0 + bkg + kk) * D + bn * 128 + bcol];
        Bs[bcol][bkg + kk] = (_Float16)v;
      }
      if (k0 + 32 < D)  // warm L2 for the next W tile (global_prefetch_b8)
        __builtin_prefetch(&W[(long)(k0 + 32 + bkg) * D + bn * 128 + bcol], 0, 1);
    }
    if (A_F16)  // async copies must land in LDS before the barrier
      asm volatile("s_wait_asynccnt 0x0" ::: "memory");
    __syncthreads();

    // ---- fragments (ds_load_b128 pairs) ----
    H16 a[2], b[4];
#pragma unroll
    for (int mi = 0; mi < 2; mi++) {
      const int row = mBase + mi * 16 + ln;
      a[mi].h[0] = *(const h8*)&As[row][8 * g];        // K: 8g..8g+7
      a[mi].h[1] = *(const h8*)&As[row][16 + 8 * g];   // K: 16+8g..
    }
#pragma unroll
    for (int ni = 0; ni < 4; ni++) {
      const int col = nBase + ni * 16 + ln;
      b[ni].h[0] = *(const h8*)&Bs[col][8 * g];        // K: 8g..8g+7
      b[ni].h[1] = *(const h8*)&Bs[col][16 + 8 * g];   // K: 16+8g..
    }
#pragma unroll
    for (int mi = 0; mi < 2; mi++)
#pragma unroll
      for (int ni = 0; ni < 4; ni++)
        acc[mi][ni] = wmma16(a[mi].v, b[ni].v, acc[mi][ni]);
    __syncthreads();
  }

  // ---- epilogue ----
  if constexpr (MODE == MODE_VT) {
    // Coalesced transposed store via LDS: two 64-column halves.
    __shared__ _Float16 Ct[64][136];  // [n][m], 272B rows (17x16B)
    for (int half = 0; half < 2; half++) {
      if ((wid & 1) == half) {  // waves whose nBase == half*64
#pragma unroll
        for (int mi = 0; mi < 2; mi++)
#pragma unroll
          for (int ni = 0; ni < 4; ni++) {
            const int gCol = bn * 128 + nBase + ni * 16 + ln;
            h8 tmp;
#pragma unroll
            for (int r = 0; r < 8; r++)
              tmp[r] = (_Float16)(acc[mi][ni][r] + bias[gCol]);
            *(h8*)&Ct[ni * 16 + ln][mBase + mi * 16 + 8 * g] = tmp;
          }
      }
      __syncthreads();
      {  // all 256 threads: 64B contiguous stores, 256B-coalesced groups
        const int n  = tid >> 2;          // 0..63
        const int mc = (tid & 3) << 5;    // 0,32,64,96
        const int gCol = bn * 128 + half * 64 + n;
        const int hh = gCol >> 6, dk = gCol & 63;
        const int bb = bm >> 3;                 // 128-row tile within one batch
        const int ssB = (bm & 7) * 128 + mc;
        _Float16* dst = ((_Float16*)outp) +
                        (((long)bb * H + hh) * DK + dk) * S + ssB;
        const _Float16* srcl = &Ct[n][mc];
#pragma unroll
        for (int i = 0; i < 4; i++)
          *(h8*)(dst + i * 8) = *(const h8*)(srcl + i * 8);
      }
      __syncthreads();
    }
  } else {
#pragma unroll
    for (int mi = 0; mi < 2; mi++)
#pragma unroll
      for (int ni = 0; ni < 4; ni++)
#pragma unroll
        for (int r = 0; r < 8; r++) {
          const int gRow = bm * 128 + mBase + mi * 16 + r + 8 * g;
          const int gCol = bn * 128 + nBase + ni * 16 + ln;
          float v = acc[mi][ni][r] + bias[gCol];
          const int bb = gRow >> 10, ss = gRow & 1023;
          const int hh = gCol >> 6, dk = gCol & 63;
          if (MODE == MODE_HEAD) {          // head-major f16 [b,h,s,dk]
            ((_Float16*)outp)[(((long)bb * H + hh) * S + ss) * DK + dk] =
                (_Float16)(v * scale);
          } else {                          // f32 y = ctx@Wo + bo + query
            ((float*)outp)[(long)gRow * D + gCol] =
                v + resid[(long)gRow * D + gCol];
          }
        }
  }
}

// ---------------------------------------------------------------------------
// Causal attention w/ state_weight bias, flash-style online softmax.
// Grid: (S/128, B*H), 256 thr (8 waves). Wave owns 16 query rows.
// ---------------------------------------------------------------------------
__global__ __launch_bounds__(256) void attn_kernel(
    const _Float16* __restrict__ qh, const _Float16* __restrict__ kh,
    const _Float16* __restrict__ vt, const float* __restrict__ sw,
    _Float16* __restrict__ ctx) {
  __shared__ _Float16 P[8][16][40];  // wave-private P (16x32 f16), padded

  const int tid  = threadIdx.x;
  const int lane = tid & 31, wid = tid >> 5;
  const int g    = lane >> 4, ln = lane & 15;
  const int bh   = blockIdx.y;            // 0..63
  const int b    = bh >> 4, h = bh & 15;
  const int q0   = blockIdx.x * 128 + wid * 16;

  const _Float16* qbase = qh + (long)bh * S * DK;
  const _Float16* kbase = kh + (long)bh * S * DK;
  const _Float16* vbase = vt + (long)bh * DK * S;
  const float*    swb   = sw + (long)h * S * S;

  // Q A-fragments straight from global (row-major [s][dk], contiguous runs)
  H16 aQ[2];
  {
    const _Float16* qrow = qbase + (long)(q0 + ln) * DK;
#pragma unroll
    for (int c = 0; c < 2; c++) {
      aQ[c].h[0] = *(const h8*)(qrow + c * 32 + 8 * g);
      aQ[c].h[1] = *(const h8*)(qrow + c * 32 + 16 + 8 * g);
    }
  }

  f8 O[4];
  float rowmax[8], rowsum[8];
#pragma unroll
  for (int ni = 0; ni < 4; ni++)
#pragma unroll
    for (int r = 0; r < 8; r++) O[ni][r] = 0.0f;
#pragma unroll
  for (int r = 0; r < 8; r++) { rowmax[r] = -1e32f; rowsum[r] = 0.0f; }

  for (int kt = 0; kt < q0 + 16; kt += 32) {   // strictly-causal key range
    float sv[2][8], cmax[8];
#pragma unroll
    for (int r = 0; r < 8; r++) cmax[r] = -3.0e38f;

#pragma unroll
    for (int j = 0; j < 2; j++) {
      // K^T B-fragments: lane = key position, dk in table order (8g, 16+8g)
      H16 bK[2];
      const _Float16* krow = kbase + (long)(kt + j * 16 + ln) * DK;
#pragma unroll
      for (int c = 0; c < 2; c++) {
        bK[c].h[0] = *(const h8*)(krow + c * 32 + 8 * g);
        bK[c].h[1] = *(const h8*)(krow + c * 32 + 16 + 8 * g);
      }
      f8 s;
#pragma unroll
      for (int r = 0; r < 8; r++) s[r] = 0.0f;
      s = wmma16(aQ[0].v, bK[0].v, s);   // Q pre-scaled by 1/sqrt(DK)
      s = wmma16(aQ[1].v, bK[1].v, s);
#pragma unroll
      for (int r = 0; r < 8; r++) {
        const int qrow = q0 + r + 8 * g;
        const int key  = kt + j * 16 + ln;
        // Unconditional (always in-bounds) load + select: no divergent branch.
        const float swv = swb[(long)qrow * S + key];
        const float v = (key < qrow) ? (s[r] + swv) : -1e32f;
        sv[j][r] = v;
        cmax[r]  = fmaxf(cmax[r], v);
      }
    }

    // online-softmax stats: reduce across the 16 lanes holding each row
    float scl[8], lsum[8];
#pragma unroll
    for (int r = 0; r < 8; r++) {
      float m = cmax[r];
      for (int off = 1; off < 16; off <<= 1) m = fmaxf(m, __shfl_xor(m, off, 16));
      const float nm = fmaxf(rowmax[r], m);
      scl[r]    = __expf(rowmax[r] - nm);
      rowmax[r] = nm;
      lsum[r]   = 0.0f;
    }
#pragma unroll
    for (int j = 0; j < 2; j++)
#pragma unroll
      for (int r = 0; r < 8; r++) {
        const float p = __expf(sv[j][r] - rowmax[r]);
        lsum[r] += p;
        P[wid][r + 8 * g][j * 16 + ln] = (_Float16)p;  // C-frag -> LDS relayout
      }
#pragma unroll
    for (int r = 0; r < 8; r++) {
      float s = lsum[r];
      for (int off = 1; off < 16; off <<= 1) s += __shfl_xor(s, off, 16);
      rowsum[r] = rowsum[r] * scl[r] + s;
#pragma unroll
      for (int ni = 0; ni < 4; ni++) O[ni][r] *= scl[r];
    }

    // DS ops are in-order within a wave; fence only the compiler.
    asm volatile("" ::: "memory");
    H16 aP;
    aP.h[0] = *(const h8*)&P[wid][ln][8 * g];
    aP.h[1] = *(const h8*)&P[wid][ln][16 + 8 * g];

#pragma unroll
    for (int ni = 0; ni < 4; ni++) {   // V^T B-fragments: lane = dk, k = keypos
      H16 bV;
      const _Float16* vrow = vbase + (long)(ni * 16 + ln) * S + kt + 8 * g;
      bV.h[0] = *(const h8*)(vrow);
      bV.h[1] = *(const h8*)(vrow + 16);
      O[ni] = wmma16(aP.v, bV.v, O[ni]);
    }
    asm volatile("" ::: "memory");
  }

  // normalize + store ctx f16 as [b,s,(h*64+dk)] (A-input of output proj)
#pragma unroll
  for (int ni = 0; ni < 4; ni++)
#pragma unroll
    for (int r = 0; r < 8; r++) {
      const int qrow = q0 + r + 8 * g;
      float v = (qrow == 0 || rowsum[r] <= 0.0f) ? 0.0f : O[ni][r] / rowsum[r];
      ctx[((long)b * S + qrow) * D + h * DK + ni * 16 + ln] = (_Float16)v;
    }
}

// ---------------------------------------------------------------------------
// LayerNorm over last dim (1024). One block per row.
// ---------------------------------------------------------------------------
__global__ __launch_bounds__(256) void ln_kernel(const float* __restrict__ y,
                                                 const float* __restrict__ gamma,
                                                 const float* __restrict__ beta,
                                                 float* __restrict__ out) {
  __shared__ float red[256];
  const int row = blockIdx.x;
  const int tid = threadIdx.x;
  const float* x = y + (long)row * D;

  float v[4], s = 0.0f, s2 = 0.0f;
#pragma unroll
  for (int j = 0; j < 4; j++) {
    v[j] = x[tid + j * 256];
    s += v[j];
    s2 += v[j] * v[j];
  }
  red[tid] = s;
  __syncthreads();
  for (int off = 128; off; off >>= 1) {
    if (tid < off) red[tid] += red[tid + off];
    __syncthreads();
  }
  const float mean = red[0] * (1.0f / D);
  __syncthreads();
  red[tid] = s2;
  __syncthreads();
  for (int off = 128; off; off >>= 1) {
    if (tid < off) red[tid] += red[tid + off];
    __syncthreads();
  }
  const float var  = red[0] * (1.0f / D) - mean * mean;
  const float rstd = rsqrtf(var + 1e-5f);
#pragma unroll
  for (int j = 0; j < 4; j++) {
    const int c = tid + j * 256;
    out[(long)row * D + c] = (v[j] - mean) * rstd * gamma[c] + beta[c];
  }
}

// ---------------------------------------------------------------------------
extern "C" void kernel_launch(void* const* d_in, const int* in_sizes, int n_in,
                              void* d_out, int out_size, void* d_ws,
                              size_t ws_size, hipStream_t stream) {
  const float* query  = (const float*)d_in[0];
  const float* key    = (const float*)d_in[1];
  const float* values = (const float*)d_in[2];
  /* d_in[3] = lens : unused by the reference computation */
  const float* sw     = (const float*)d_in[4];   // [1,H,S,S]
  const float* Wq     = (const float*)d_in[5];
  const float* bq     = (const float*)d_in[6];
  const float* Wv     = (const float*)d_in[7];
  const float* bv     = (const float*)d_in[8];
  const float* Wo     = (const float*)d_in[9];
  const float* bo     = (const float*)d_in[10];
  const float* gamma  = (const float*)d_in[11];
  const float* beta   = (const float*)d_in[12];
  float* out = (float*)d_out;

  // workspace carve-up (48 MB total): qh/kh/vt/ctx f16 (8 MB each) + y f32 (16 MB)
  char* ws = (char*)d_ws;
  _Float16* qh  = (_Float16*)(ws);
  _Float16* kh  = (_Float16*)(ws + (size_t)8  * 1024 * 1024);
  _Float16* vt  = (_Float16*)(ws + (size_t)16 * 1024 * 1024);
  _Float16* ctx = (_Float16*)(ws + (size_t)24 * 1024 * 1024);
  float*    y   = (float*)   (ws + (size_t)32 * 1024 * 1024);

  const dim3 gg(D / 128, M / 128), gb(256);
  // Q = query@Wq+bq, pre-scaled 1/sqrt(DK); K shares Wq (kq_same)
  gemm_wmma<MODE_HEAD, false><<<gg, gb, 0, stream>>>(query,  Wq, bq, nullptr, qh, 0.125f);
  gemm_wmma<MODE_HEAD, false><<<gg, gb, 0, stream>>>(key,    Wq, bq, nullptr, kh, 1.0f);
  gemm_wmma<MODE_VT,   false><<<gg, gb, 0, stream>>>(values, Wv, bv, nullptr, vt, 1.0f);
  attn_kernel<<<dim3(S / 128, Bb * H), 256, 0, stream>>>(qh, kh, vt, sw, ctx);
  gemm_wmma<MODE_Y, true><<<gg, gb, 0, stream>>>(ctx, Wo, bo, query, y, 1.0f);
  ln_kernel<<<dim3(M), 256, 0, stream>>>(y, gamma, beta, out);

  (void)in_sizes; (void)n_in; (void)out_size; (void)ws_size;
}
